// PMM_45578192945382
// MI455X (gfx1250) — compile-verified, matrix operand order checked by reading, not run on hardware
//
#include <hip/hip_runtime.h>
#include <hip/hip_bf16.h>
#include <math.h>

// ---------------------------------------------------------------------------
// Types for CDNA5 WMMA (wave32, 16x16x32 f16 -> f32)
// ---------------------------------------------------------------------------
typedef _Float16 half2v __attribute__((ext_vector_type(2)));
typedef _Float16 half8 __attribute__((ext_vector_type(8)));
typedef _Float16 v16h  __attribute__((ext_vector_type(16)));
typedef float    v8f   __attribute__((ext_vector_type(8)));
typedef float    f32x4 __attribute__((ext_vector_type(4)));

#define BATCH 64
#define NTOK  256
#define DIM   512
#define ROWS  (BATCH * NTOK)   // 16384 token rows

__device__ __forceinline__ float geluf(float x) {
  return 0.5f * x * (1.0f + erff(x * 0.70710678118654752f));
}

__device__ __forceinline__ v8f wmma32(v16h a, v16h b, v8f c) {
  // D = A(16x32) x B(32x16) + C, f32 accumulate
  return __builtin_amdgcn_wmma_f32_16x16x32_f16(false, a, false, b, (short)0, c,
                                                false, false);
}

// Fragment loader for BOTH A and B operands of 16x16x32 f16 WMMA from a
// row-major [16 x >=32] tile (activations rows for A, W[n,:] rows for B).
// ISA 16-bit A layout: lane<16 -> row=lane,  K = {0..7, 16..23}
//                      lane>=16 -> row=lane-16, K = {8..15, 24..31}
// ldHalves must keep 16-byte alignment for ko offsets (multiples of 8 halves).
__device__ __forceinline__ v16h load_frag(const _Float16* p, int ldHalves) {
  const int lane = threadIdx.x & 31;
  const int r    = lane & 15;
  const int ko   = (lane >> 4) << 3;
  const _Float16* q = p + r * ldHalves + ko;
  v16h f;
  *(half8*)&f          = *(const half8*)(q);
  *(((half8*)&f) + 1)  = *(const half8*)(q + 16);
  return f;
}

// ---------------------------------------------------------------------------
// Generic tiled GEMM:  Y[M,N] = A[M,K](f16) @ W[N,K]^T(f32, cvt->f16 in LDS)
// Block: 128(M) x 64(N), K staged 32 at a time. 256 threads = 8 waves;
// wave w owns M-rows [w*16, w*16+16) across all 4 N-subtiles.
// Epilogues (template EPI):
//   0: outF = acc + bias[col]                  (patch embed, pre-LN)
//   1: scatter QKV to [B,H,N,D] f16, Q scaled by 1/8
//   2: outF = acc + res                        (attn out proj + residual)
//   3: outH = f16(gelu(acc + bias))            (ff1)
//   4: outF = acc + bias + res                 (ff2 + residual)
// ---------------------------------------------------------------------------
template <int EPI>
__global__ __launch_bounds__(256) void gemm_f16(
    const _Float16* __restrict__ A, const float* __restrict__ W,
    const float* __restrict__ bias, const float* __restrict__ res,
    float* __restrict__ outF, _Float16* __restrict__ outH,
    _Float16* __restrict__ oq, _Float16* __restrict__ okk,
    _Float16* __restrict__ ov, int K, int N) {
  __shared__ _Float16 sA[128 * 32];
  __shared__ _Float16 sB[64 * 32];
  const int m0 = blockIdx.x * 128;
  const int n0 = blockIdx.y * 64;
  const int t = threadIdx.x;
  const int wave = t >> 5, lane = t & 31;
  v8f acc[4] = {};

  const int aRow = t >> 1, aK0 = (t & 1) << 4;  // 16 halves / thread
  const int bRow = t >> 2, bK0 = (t & 3) << 3;  // 8 floats / thread

  for (int ks = 0; ks < K; ks += 32) {
    const _Float16* ap = A + (size_t)(m0 + aRow) * K + ks + aK0;
    *(half8*)&sA[aRow * 32 + aK0]     = *(const half8*)(ap);
    *(half8*)&sA[aRow * 32 + aK0 + 8] = *(const half8*)(ap + 8);
    const float* wp = W + (size_t)(n0 + bRow) * K + ks + bK0;
    {
      f32x4 w0 = *(const f32x4*)(wp);
      f32x4 w1 = *(const f32x4*)(wp + 4);
      half8 hb;
#pragma unroll
      for (int i = 0; i < 4; ++i) {
        hb[i]     = (_Float16)w0[i];
        hb[i + 4] = (_Float16)w1[i];
      }
      *(half8*)&sB[bRow * 32 + bK0] = hb;
    }
    if (ks + 32 < K) {  // gfx1250 global_prefetch of next K-slice
      __builtin_prefetch(ap + 32, 0, 3);
      __builtin_prefetch(wp + 32, 0, 3);
    }
    __syncthreads();
    v16h a = load_frag(&sA[wave * 16 * 32], 32);
#pragma unroll
    for (int j = 0; j < 4; ++j) {
      v16h b = load_frag(&sB[j * 16 * 32], 32);
      acc[j] = wmma32(a, b, acc[j]);
    }
    __syncthreads();
  }

  // C layout: reg r, lane l -> row = r + (l>=16 ? 8 : 0), col = l&15
  const int rbase = m0 + wave * 16 + ((lane >> 4) << 3);
  const int cl = lane & 15;
#pragma unroll
  for (int j = 0; j < 4; ++j) {
    const int col = n0 + j * 16 + cl;
#pragma unroll
    for (int r = 0; r < 8; ++r) {
      const int row = rbase + r;
      const float v = acc[j][r];
      if (EPI == 0) {
        outF[(size_t)row * N + col] = v + bias[col];
      } else if (EPI == 1) {
        const int b_ = row >> 8, tok = row & 255;
        const int which = col >> 9;      // 0=q 1=k 2=v
        const int hh = (col >> 6) & 7;
        const int dd = col & 63;
        const size_t dst = (((size_t)(b_ * 8 + hh)) * 256 + tok) * 64 + dd;
        const float sv = (which == 0) ? v * 0.125f : v;  // fold DHEAD^-0.5 into Q
        _Float16* dp = (which == 0) ? oq : (which == 1) ? okk : ov;
        dp[dst] = (_Float16)sv;
      } else if (EPI == 2) {
        outF[(size_t)row * N + col] = v + res[(size_t)row * N + col];
      } else if (EPI == 3) {
        outH[(size_t)row * N + col] = (_Float16)geluf(v + bias[col]);
      } else if (EPI == 4) {
        outF[(size_t)row * N + col] =
            v + bias[col] + res[(size_t)row * N + col];
      }
    }
  }
}

// ---------------------------------------------------------------------------
// 3x3 SAME conv as implicit GEMM with WMMA.
// M = B*32*32 spatial (x fastest), N = Cout, K = 9*Cin (kpos-major).
// Branchless im2col: clamped (always-legal) address + per-element select,
// packed into half8 registers -> two ds_store_b128 per thread.
// ---------------------------------------------------------------------------
__global__ __launch_bounds__(256) void conv3x3_wmma(
    const float* __restrict__ in, const float* __restrict__ wgt,
    const float* __restrict__ bias, float* __restrict__ out, int Cin,
    int Cout) {
  __shared__ _Float16 sA[128 * 32];
  __shared__ _Float16 sB[64 * 32];
  const int m0 = blockIdx.x * 128;  // 128 spatial positions (4 image rows)
  const int n0 = blockIdx.y * 64;
  const int t = threadIdx.x;
  const int wave = t >> 5, lane = t & 31;
  v8f acc[4] = {};

  const int aRow = t >> 1, aK0 = (t & 1) << 4;
  const int m = m0 + aRow;
  const int bimg = m >> 10;
  const int y = (m >> 5) & 31;
  const int x = m & 31;
  const int bRow = t >> 2, bK0 = (t & 3) << 3;

  for (int kp = 0; kp < 9; ++kp) {
    const int yy = y + kp / 3 - 1;
    const int xx = x + kp % 3 - 1;
    const bool valid = ((unsigned)yy < 32u) && ((unsigned)xx < 32u);
    const int yc = yy < 0 ? 0 : (yy > 31 ? 31 : yy);
    const int xc = xx < 0 ? 0 : (xx > 31 ? 31 : xx);
    for (int c0 = 0; c0 < Cin; c0 += 32) {
      const size_t pbase =
          (((size_t)bimg * Cin + c0 + aK0) << 10) + (yc << 5) + xc;
      half8 h0, h1;
#pragma unroll
      for (int i = 0; i < 8; ++i) {
        const float v0 = in[pbase + ((size_t)i << 10)];
        h0[i] = (_Float16)(valid ? v0 : 0.0f);
      }
#pragma unroll
      for (int i = 0; i < 8; ++i) {
        const float v1 = in[pbase + ((size_t)(8 + i) << 10)];
        h1[i] = (_Float16)(valid ? v1 : 0.0f);
      }
      *(half8*)&sA[aRow * 32 + aK0]     = h0;
      *(half8*)&sA[aRow * 32 + aK0 + 8] = h1;
      {
        half8 hb;
#pragma unroll
        for (int i = 0; i < 8; ++i)
          hb[i] = (_Float16)
              wgt[((size_t)(n0 + bRow) * Cin + c0 + bK0 + i) * 9 + kp];
        *(half8*)&sB[bRow * 32 + bK0] = hb;
      }
      __syncthreads();
      v16h a = load_frag(&sA[wave * 16 * 32], 32);
#pragma unroll
      for (int j = 0; j < 4; ++j) {
        v16h b = load_frag(&sB[j * 16 * 32], 32);
        acc[j] = wmma32(a, b, acc[j]);
      }
      __syncthreads();
    }
  }

  const int rbase = wave * 16 + ((lane >> 4) << 3);
  const int cl = lane & 15;
#pragma unroll
  for (int j = 0; j < 4; ++j) {
    const int co = n0 + j * 16 + cl;
    const float bv = bias[co];
#pragma unroll
    for (int r = 0; r < 8; ++r) {
      const int mm = m0 + rbase + r;
      const int bb = mm >> 10, pix = mm & 1023;
      out[(((size_t)bb * Cout + co) << 10) + pix] = acc[j][r] + bv;
    }
  }
}

// ---------------------------------------------------------------------------
// Attention: one block per (b,h). Everything LDS-resident (~203KB of the
// 320KB/WGP CDNA5 LDS): K rows, V^T, and the full 256x256 score matrix.
// QK^T via WMMA -> per-row softmax (pose bias folded in) -> P@V via WMMA.
// Q is pre-scaled by 1/8 at QKV-GEMM epilogue; beta folded into staged pb.
// ---------------------------------------------------------------------------
#define KS_H (256 * 64)
#define VT_H (64 * 264)
#define SC_H (256 * 264)
#define ATTN_SMEM ((size_t)(KS_H + VT_H + SC_H) * 2 + 256 * 4)

__global__ __launch_bounds__(256) void attn_kernel(
    const _Float16* __restrict__ q, const _Float16* __restrict__ k,
    const _Float16* __restrict__ v, const float* __restrict__ pb,
    const float* __restrict__ beta, int layer, _Float16* __restrict__ o) {
  extern __shared__ _Float16 smem[];
  _Float16* ksm = smem;            // [256][64]
  _Float16* vT  = smem + KS_H;     // [64][264]  (padded: 528B rows)
  _Float16* sc  = vT + VT_H;       // [256][264] scores/probs
  float* pbs = (float*)(sc + SC_H);

  const int bh = blockIdx.x;
  const int b = bh >> 3, h = bh & 7;
  const int t = threadIdx.x, wave = t >> 5, lane = t & 31;
  const size_t base = (size_t)bh * 256 * 64;
  const float bet = beta[layer];

  {  // stage K (row-major)
    const _Float16* kp = k + base + (size_t)t * 64;
#pragma unroll
    for (int i = 0; i < 8; ++i)
      *(half8*)&ksm[t * 64 + i * 8] = *(const half8*)(kp + i * 8);
    pbs[t] = bet * pb[b * 256 + t];  // beta folded in
  }
  {  // stage V^T: thread handles token pair (tok2, tok2+1) x 32 dims
    const int tok2 = (t & 127) << 1;
    const int dh = (t >> 7) << 5;  // 0 or 32
    const _Float16* vp0 = v + base + (size_t)tok2 * 64 + dh;
#pragma unroll
    for (int i = 0; i < 4; ++i) {
      half8 a0 = *(const half8*)(vp0 + i * 8);
      half8 a1 = *(const half8*)(vp0 + 64 + i * 8);
#pragma unroll
      for (int jj = 0; jj < 8; ++jj) {
        const int d = dh + i * 8 + jj;
        half2v pr;
        pr[0] = a0[jj];
        pr[1] = a1[jj];
        *(half2v*)&vT[d * 264 + tok2] = pr;
      }
    }
  }

  // Q fragments straight from global (row-major, stride 64 halves)
  v16h qa[2][2];
#pragma unroll
  for (int mi = 0; mi < 2; ++mi)
#pragma unroll
    for (int s = 0; s < 2; ++s)
      qa[mi][s] =
          load_frag(q + base + (size_t)(wave * 32 + mi * 16) * 64 + s * 32, 64);
  __syncthreads();

  // dots = (Q*scale) @ K^T : wave w owns rows [32w, 32w+32)
  for (int j = 0; j < 16; ++j) {
    v16h b0 = load_frag(&ksm[(j * 16) * 64], 64);
    v16h b1 = load_frag(&ksm[(j * 16) * 64 + 32], 64);
#pragma unroll
    for (int mi = 0; mi < 2; ++mi) {
      v8f c = {};
      c = wmma32(qa[mi][0], b0, c);
      c = wmma32(qa[mi][1], b1, c);
      const int rb = wave * 32 + mi * 16 + ((lane >> 4) << 3);
      const int col = j * 16 + (lane & 15);
#pragma unroll
      for (int r = 0; r < 8; ++r) sc[(rb + r) * 264 + col] = (_Float16)c[r];
    }
  }
  __syncthreads();

  {  // softmax: thread t owns row t; bias = beta*pb[row] + beta*pb[col]
    const float rowb = pbs[t];
    float mx = -1e30f;
    for (int n = 0; n < 256; ++n) {
      float d = (float)sc[t * 264 + n] + rowb + pbs[n];
      mx = fmaxf(mx, d);
    }
    float s = 0.f;
    for (int n = 0; n < 256; ++n) {
      float d = (float)sc[t * 264 + n] + rowb + pbs[n];
      float e = __expf(d - mx);
      s += e;
      sc[t * 264 + n] = (_Float16)e;
    }
    const float inv = 1.0f / s;
    for (int n = 0; n < 256; ++n)
      sc[t * 264 + n] = (_Float16)((float)sc[t * 264 + n] * inv);
  }
  __syncthreads();

  // O = P @ V : K=256 in 8 slices of 32
#pragma unroll
  for (int mi = 0; mi < 2; ++mi) {
    v8f acc[4] = {};
    for (int s = 0; s < 8; ++s) {
      v16h a = load_frag(&sc[(wave * 32 + mi * 16) * 264 + s * 32], 264);
#pragma unroll
      for (int j = 0; j < 4; ++j) {
        v16h bb = load_frag(&vT[(j * 16) * 264 + s * 32], 264);
        acc[j] = wmma32(a, bb, acc[j]);
      }
    }
    const int rb = wave * 32 + mi * 16 + ((lane >> 4) << 3);
#pragma unroll
    for (int j = 0; j < 4; ++j) {
      const int d = j * 16 + (lane & 15);
#pragma unroll
      for (int r = 0; r < 8; ++r) {
        const int tok = rb + r;
        o[((size_t)(b * 256 + tok)) * 512 + h * 64 + d] = (_Float16)acc[j][r];
      }
    }
  }
}

// ---------------------------------------------------------------------------
// LayerNorm over 512: one wave32 per row, 16 elems/lane, shfl_xor reductions.
// outH: f16 (GEMM A input); outF: f32 (+optional posemb) for the master x.
// ---------------------------------------------------------------------------
__global__ __launch_bounds__(256) void ln_kernel(
    const float* __restrict__ in, const float* __restrict__ g,
    const float* __restrict__ bg, _Float16* __restrict__ outH,
    float* __restrict__ outF, const float* __restrict__ pos, int M) {
  const int row = blockIdx.x * 8 + (threadIdx.x >> 5);
  const int lane = threadIdx.x & 31;
  if (row >= M) return;
  const float* p = in + (size_t)row * 512 + lane * 16;
  float vals[16];
#pragma unroll
  for (int i = 0; i < 16; ++i) vals[i] = p[i];
  float s = 0.f;
#pragma unroll
  for (int i = 0; i < 16; ++i) s += vals[i];
  for (int off = 16; off > 0; off >>= 1) s += __shfl_xor(s, off, 32);
  const float mean = s * (1.0f / 512.0f);
  float vs = 0.f;
#pragma unroll
  for (int i = 0; i < 16; ++i) {
    float d = vals[i] - mean;
    vs += d * d;
  }
  for (int off = 16; off > 0; off >>= 1) vs += __shfl_xor(vs, off, 32);
  const float inv = rsqrtf(vs * (1.0f / 512.0f) + 1e-5f);
  const int j0 = lane * 16;
#pragma unroll
  for (int i = 0; i < 16; ++i) {
    float ov = (vals[i] - mean) * inv * g[j0 + i] + bg[j0 + i];
    if (pos) ov += pos[(size_t)(row & 255) * 512 + j0 + i];
    if (outH) outH[(size_t)row * 512 + j0 + i] = (_Float16)ov;
    if (outF) outF[(size_t)row * 512 + j0 + i] = ov;
  }
}

// LN over gathered patch vectors: conv1[B,128,32,32] -> [B,256,512] order
// (p1,p2,c), normalized, written f16.
__global__ __launch_bounds__(256) void ln_patch_kernel(
    const float* __restrict__ conv1, const float* __restrict__ g,
    const float* __restrict__ bg, _Float16* __restrict__ outH) {
  const int row = blockIdx.x * 8 + (threadIdx.x >> 5);  // b*256+tok
  const int lane = threadIdx.x & 31;
  const int b = row >> 8, tok = row & 255;
  const int hh = tok >> 4, ww = tok & 15;
  float vals[16];
#pragma unroll
  for (int i = 0; i < 16; ++i) {
    const int j = lane * 16 + i;
    const int p1 = j >> 8, p2 = (j >> 7) & 1, c = j & 127;
    vals[i] = conv1[(((size_t)(b * 128 + c)) << 10) +
                    (((hh << 1) + p1) << 5) + ((ww << 1) + p2)];
  }
  float s = 0.f;
#pragma unroll
  for (int i = 0; i < 16; ++i) s += vals[i];
  for (int off = 16; off > 0; off >>= 1) s += __shfl_xor(s, off, 32);
  const float mean = s * (1.0f / 512.0f);
  float vs = 0.f;
#pragma unroll
  for (int i = 0; i < 16; ++i) {
    float d = vals[i] - mean;
    vs += d * d;
  }
  for (int off = 16; off > 0; off >>= 1) vs += __shfl_xor(vs, off, 32);
  const float inv = rsqrtf(vs * (1.0f / 512.0f) + 1e-5f);
  const int j0 = lane * 16;
#pragma unroll
  for (int i = 0; i < 16; ++i)
    outH[(size_t)row * 512 + j0 + i] =
        (_Float16)((vals[i] - mean) * inv * g[j0 + i] + bg[j0 + i]);
}

// sincos 2d positional embedding [256,512]
__global__ __launch_bounds__(256) void posemb_kernel(float* __restrict__ pos) {
  const int idx = blockIdx.x * 256 + threadIdx.x;  // < 131072
  const int tok = idx >> 9, j = idx & 511;
  const int yy = tok >> 4, xx = tok & 15;
  const int quad = j >> 7, j0 = j & 127;
  const float om = powf(10000.0f, -(float)j0 / 127.0f);
  float val;
  if (quad == 0)      val = sinf((float)xx * om);
  else if (quad == 1) val = cosf((float)xx * om);
  else if (quad == 2) val = sinf((float)yy * om);
  else                val = cosf((float)yy * om);
  pos[idx] = val;
}

// mask pool->binarize->affine, then tiny MLP 1->64->32->1 + sigmoid => pb[B,N]
// (row-max normalization is identity here since pw in {0,1}).
__global__ __launch_bounds__(256) void pose_kernel(
    const float* __restrict__ mask, const float* __restrict__ w1,
    const float* __restrict__ b1, const float* __restrict__ w2,
    const float* __restrict__ b2, const float* __restrict__ w3,
    const float* __restrict__ b3, float* __restrict__ pb) {
  const int b = blockIdx.x, t = threadIdx.x;
  const int hh = t >> 4, ww = t & 15;
  float s = 0.f;
  for (int ky = 0; ky < 4; ++ky)
    for (int kx = 0; kx < 4; ++kx)
      s += mask[(size_t)b * 4096 + ((hh * 4 + ky) << 6) + (ww * 4 + kx)];
  const float pw = (s > 0.f ? 1.0f : 0.0f) * 0.98f + 0.02f;
  float h1[64];
  for (int j = 0; j < 64; ++j) h1[j] = geluf(pw * w1[j] + b1[j]);
  float acc3 = b3[0];
  for (int j = 0; j < 32; ++j) {
    float a = b2[j];
    for (int i = 0; i < 64; ++i) a += h1[i] * w2[j * 64 + i];
    acc3 += geluf(a) * w3[j];
  }
  pb[b * 256 + t] = 1.0f / (1.0f + __expf(-acc3));
}

// [B,256,512] tokens -> [B,128,32,32] NCHW f32 for the decode conv
__global__ __launch_bounds__(256) void rearrange_kernel(
    const float* __restrict__ xb, float* __restrict__ dec) {
  const int idx = blockIdx.x * 256 + threadIdx.x;  // < 64*128*1024
  const int b = idx >> 17;
  const int r = idx & 131071;
  const int c = r >> 10;
  const int y = (r >> 5) & 31;
  const int x = r & 31;
  const int tok = ((y >> 1) << 4) + (x >> 1);
  const int pd = (((y & 1) << 1) + (x & 1)) * 128 + c;
  dec[idx] = xb[((size_t)(b * 256 + tok)) * 512 + pd];
}

// ---------------------------------------------------------------------------
extern "C" void kernel_launch(void* const* d_in, const int* in_sizes, int n_in,
                              void* d_out, int out_size, void* d_ws,
                              size_t ws_size, hipStream_t stream) {
  (void)in_sizes; (void)n_in; (void)out_size; (void)ws_size;
  const float* img       = (const float*)d_in[0];
  const float* maskp     = (const float*)d_in[1];
  const float* conv_in_w = (const float*)d_in[2];
  const float* conv_in_b = (const float*)d_in[3];
  const float* ln_p1_g   = (const float*)d_in[4];
  const float* ln_p1_b   = (const float*)d_in[5];
  const float* patch_w   = (const float*)d_in[6];
  const float* patch_b   = (const float*)d_in[7];
  const float* ln_p2_g   = (const float*)d_in[8];
  const float* ln_p2_b   = (const float*)d_in[9];
  const float* ln_a_g    = (const float*)d_in[10];
  const float* ln_a_b    = (const float*)d_in[11];
  const float* qkv_w     = (const float*)d_in[12];
  const float* out_w     = (const float*)d_in[13];
  const float* beta      = (const float*)d_in[14];
  const float* ln_f_g    = (const float*)d_in[15];
  const float* ln_f_b    = (const float*)d_in[16];
  const float* ff1_w     = (const float*)d_in[17];
  const float* ff1_b     = (const float*)d_in[18];
  const float* ff2_w     = (const float*)d_in[19];
  const float* ff2_b     = (const float*)d_in[20];
  const float* pm_w1     = (const float*)d_in[21];
  const float* pm_b1     = (const float*)d_in[22];
  const float* pm_w2     = (const float*)d_in[23];
  const float* pm_b2     = (const float*)d_in[24];
  const float* pm_w3     = (const float*)d_in[25];
  const float* pm_b3     = (const float*)d_in[26];
  const float* conv_out_w = (const float*)d_in[27];
  const float* conv_out_b = (const float*)d_in[28];
  float* out = (float*)d_out;

  // workspace carve
  char* w = (char*)d_ws;
  auto carve = [&](size_t bytes) {
    char* p = w;
    w += (bytes + 255) & ~(size_t)255;
    return p;
  };
  float*    conv1  = (float*)carve((size_t)BATCH * 128 * 1024 * 4);  // also decode input
  _Float16* patchA = (_Float16*)carve((size_t)ROWS * DIM * 2);       // also attn output
  float*    tmp    = (float*)carve((size_t)ROWS * DIM * 4);
  float*    xbuf   = (float*)carve((size_t)ROWS * DIM * 4);          // fp32 master
  _Float16* xn     = (_Float16*)carve((size_t)ROWS * DIM * 2);
  _Float16* qb     = (_Float16*)carve((size_t)ROWS * DIM * 2);       // [B,H,N,64]
  _Float16* kb     = (_Float16*)carve((size_t)ROWS * DIM * 2);
  _Float16* vb     = (_Float16*)carve((size_t)ROWS * DIM * 2);
  _Float16* ffh    = (_Float16*)carve((size_t)ROWS * DIM * 2);
  float*    pos    = (float*)carve((size_t)NTOK * DIM * 4);
  float*    pbw    = (float*)carve((size_t)BATCH * NTOK * 4);
  _Float16* ob     = patchA;  // reuse after patch GEMM

  (void)hipFuncSetAttribute((const void*)attn_kernel,
                            hipFuncAttributeMaxDynamicSharedMemorySize,
                            (int)ATTN_SMEM);

  // posemb + pose-bias MLP (independent of main path)
  posemb_kernel<<<512, 256, 0, stream>>>(pos);
  pose_kernel<<<64, 256, 0, stream>>>(maskp, pm_w1, pm_b1, pm_w2, pm_b2, pm_w3,
                                      pm_b3, pbw);

  // patch embed: conv(320->128) -> gather+LN -> GEMM(+bias) -> LN(+posemb)
  conv3x3_wmma<<<dim3(512, 2), 256, 0, stream>>>(img, conv_in_w, conv_in_b,
                                                 conv1, 320, 128);
  ln_patch_kernel<<<ROWS / 8, 256, 0, stream>>>(conv1, ln_p1_g, ln_p1_b,
                                                patchA);
  gemm_f16<0><<<dim3(ROWS / 128, DIM / 64), 256, 0, stream>>>(
      patchA, patch_w, patch_b, nullptr, tmp, nullptr, nullptr, nullptr,
      nullptr, DIM, DIM);
  ln_kernel<<<ROWS / 8, 256, 0, stream>>>(tmp, ln_p2_g, ln_p2_b, nullptr, xbuf,
                                          pos, ROWS);

  for (int L = 0; L < 6; ++L) {
    ln_kernel<<<ROWS / 8, 256, 0, stream>>>(xbuf, ln_a_g + L * DIM,
                                            ln_a_b + L * DIM, xn, nullptr,
                                            nullptr, ROWS);
    gemm_f16<1><<<dim3(ROWS / 128, 1536 / 64), 256, 0, stream>>>(
        xn, qkv_w + (size_t)L * 1536 * DIM, nullptr, nullptr, nullptr, nullptr,
        qb, kb, vb, DIM, 1536);
    attn_kernel<<<BATCH * 8, 256, ATTN_SMEM, stream>>>(qb, kb, vb, pbw, beta,
                                                       L, ob);
    gemm_f16<2><<<dim3(ROWS / 128, DIM / 64), 256, 0, stream>>>(
        ob, out_w + (size_t)L * DIM * DIM, nullptr, xbuf, xbuf, nullptr,
        nullptr, nullptr, nullptr, DIM, DIM);
    ln_kernel<<<ROWS / 8, 256, 0, stream>>>(xbuf, ln_f_g + L * DIM,
                                            ln_f_b + L * DIM, xn, nullptr,
                                            nullptr, ROWS);
    gemm_f16<3><<<dim3(ROWS / 128, DIM / 64), 256, 0, stream>>>(
        xn, ff1_w + (size_t)L * DIM * DIM, ff1_b + L * DIM, nullptr, nullptr,
        ffh, nullptr, nullptr, nullptr, DIM, DIM);
    gemm_f16<4><<<dim3(ROWS / 128, DIM / 64), 256, 0, stream>>>(
        ffh, ff2_w + (size_t)L * DIM * DIM, ff2_b + L * DIM, xbuf, xbuf,
        nullptr, nullptr, nullptr, nullptr, DIM, DIM);
  }

  // decode: tokens -> NCHW, conv(128->320) straight into d_out
  rearrange_kernel<<<(BATCH * 128 * 1024) / 256, 256, 0, stream>>>(xbuf, conv1);
  conv3x3_wmma<<<dim3(512, 5), 256, 0, stream>>>(conv1, conv_out_w, conv_out_b,
                                                 out, 128, 320);
}
